// MultiHeadDilatedState_16329465660008
// MI455X (gfx1250) — matrix operand order repaired and checked by reading, not compiled, not run on hardware
//
#include <hip/hip_runtime.h>
#include <hip/hip_bf16.h>

// ---------------------------------------------------------------------------
// Types for CDNA5 WMMA / TDM (wave32)
// ---------------------------------------------------------------------------
typedef __attribute__((ext_vector_type(16))) __bf16        v16bf;
typedef __attribute__((ext_vector_type(8)))  __bf16        v8bf;
typedef __attribute__((ext_vector_type(8)))  float         v8f;
typedef __attribute__((ext_vector_type(4)))  unsigned int  u32x4;
typedef __attribute__((ext_vector_type(8)))  int           i32x8;
typedef __attribute__((ext_vector_type(4)))  int           i32x4;

#define NHEADS 12
#define HID    768
#define HDIM   64
#define SEQ    4096
#define BATCH  4
#define MROWS  (BATCH*SEQ)      // 16384
#define MEMDIM 128

__device__ __constant__ int c_dil[NHEADS][3] = {
  {1,2,4},{1,1,1},{4,8,16},{8,16,32},{32,64,128},{64,128,256},
  {256,512,1024},{1,100,200},{1,500,1000},{1,1024,2048},{3,9,27},{5,25,125}
};

__device__ __forceinline__ float sigmoidf(float x) { return 1.0f / (1.0f + __expf(-x)); }

// ---------------------------------------------------------------------------
// WMMA fragment loaders (wave32 layouts per CDNA5 ISA 7.12.2)
// A (16x32 bf16): lane L -> row = L&15, half = L>>4;
//   elems 0..7 = K(half*8+0..7), elems 8..15 = K(16+half*8+0..7)  -> two 16B loads
// B (32x16 bf16 stored [n][k]): lane L -> col = L&15, half = L>>4;
//   elems 0..7 = K(half*8+0..7), elems 8..15 = K(16+half*8+0..7)  -> two 16B loads
// C/D f32: lane L -> col = L&15; vgpr r -> row = r + 8*(L>>4)
// LDS strides are multiples of 8 elems so all v8bf loads stay 16B aligned.
// ---------------------------------------------------------------------------
__device__ __forceinline__ v16bf load_a_frag(const __bf16* A, int lda, int r0, int k0, int lane) {
  const __bf16* p = A + (size_t)(r0 + (lane & 15)) * lda + k0 + ((lane >> 4) << 3);
  v8bf lo = *(const v8bf*)(p);
  v8bf hi = *(const v8bf*)(p + 16);
  v16bf r;
#pragma unroll
  for (int i = 0; i < 8; ++i) { r[i] = lo[i]; r[i + 8] = hi[i]; }
  return r;
}

__device__ __forceinline__ v16bf load_b_frag(const __bf16* Bm, int ldb, int n0, int k0, int lane) {
  const __bf16* p = Bm + (size_t)(n0 + (lane & 15)) * ldb + k0 + ((lane >> 4) << 3);
  v8bf lo = *(const v8bf*)(p);
  v8bf hi = *(const v8bf*)(p + 16);
  v16bf r;
#pragma unroll
  for (int i = 0; i < 8; ++i) { r[i] = lo[i]; r[i + 8] = hi[i]; }
  return r;
}

// Wave-level LDS GEMM helper used by the neural-memory kernel.
template <typename Epi>
__device__ __forceinline__ void wave_gemm(const __bf16* A, int lda, const __bf16* Bm, int ldb,
                                          int Mt, int Nt, int Kt, int wave, int lane, Epi epi) {
  const int tn_ = Nt >> 4;
  const int nT  = (Mt >> 4) * tn_;
  for (int t = wave; t < nT; t += 8) {
    const int tm = t / tn_, tn = t % tn_;
    v8f acc = {};
    for (int k0 = 0; k0 < Kt; k0 += 32) {
      v16bf a  = load_a_frag(A, lda, tm << 4, k0, lane);
      v16bf bb = load_b_frag(Bm, ldb, tn << 4, k0, lane);
      acc = __builtin_amdgcn_wmma_f32_16x16x32_bf16(false, a, false, bb, (short)0, acc, false, false);
    }
    const int col = (tn << 4) + (lane & 15);
    const int rb  = (tm << 4) + ((lane >> 4) << 3);
#pragma unroll
    for (int r = 0; r < 8; ++r) epi(rb + r, col, acc[r]);
  }
}

// ---------------------------------------------------------------------------
// Tensor Data Mover: DMA a 2D bf16 tile [rows x 64] (row stride = K elems) from
// global into LDS, padding each 128B row with 16B -> LDS row stride 72 elems.
// D# bitfields per CDNA5 ISA 8.3/8.4. Issued by one wave; tracked by TENSORcnt.
// ---------------------------------------------------------------------------
__device__ __forceinline__ unsigned int lds_off(const void* p) {
  return (unsigned int)(uintptr_t)p;   // low 32 bits of flat addr = LDS byte offset
}

__device__ __forceinline__ void tdm_load_tile64(unsigned int lds_addr, const __bf16* gptr,
                                                unsigned int rows, unsigned int row_stride_elems) {
  unsigned long long ga = (unsigned long long)(uintptr_t)gptr;
  u32x4 g0;
  g0[0] = 1u;                                            // count=1, user mode
  g0[1] = lds_addr;                                      // lds_addr
  g0[2] = (unsigned int)ga;                              // global_addr[31:0]
  g0[3] = (unsigned int)((ga >> 32) & 0x1FFFFFFull)      // global_addr[56:32]
        | (2u << 30);                                    // type=2 (image)
  i32x8 g1;
  g1[0] = (int)((1u << 16)      // data_size = 2 bytes
              | (1u << 20)      // pad_enable
              | (4u << 22)      // pad_interval: 8<<4 = 128B between pads
              | (3u << 25));    // pad_amount: 4 DWORDs = 16B
  const unsigned int td0 = 64;            // tensor_dim0 (elems) == tile width
  const unsigned int td1 = rows;          // tensor_dim1 == tile rows
  g1[1] = (int)((td0 & 0xFFFFu) << 16);                       // tensor_dim0[15:0] @ bit48
  g1[2] = (int)((td0 >> 16) | ((td1 & 0xFFFFu) << 16));       // dim0 hi | dim1 lo
  g1[3] = (int)((td1 >> 16) | (64u << 16));                   // dim1 hi | tile_dim0=64
  g1[4] = (int)(rows & 0xFFFFu);                              // tile_dim1=rows, tile_dim2=0
  g1[5] = (int)row_stride_elems;                              // tensor_dim0_stride lo32
  g1[6] = 0;                                                  // stride hi | dim1_stride lo
  g1[7] = 0;
  i32x4 z4 = (i32x4){0, 0, 0, 0};
#if __clang_major__ >= 23
  i32x8 z8 = (i32x8){0, 0, 0, 0, 0, 0, 0, 0};
  __builtin_amdgcn_tensor_load_to_lds(g0, g1, z4, z4, z8, 0);
#else
  __builtin_amdgcn_tensor_load_to_lds(g0, g1, z4, z4, 0);
#endif
}

// ---------------------------------------------------------------------------
// Main GEMM: C[M,N] = A[M,K] @ W[N,K]^T (+bias); bf16 in, f32 out.
// 256 threads = 8 wave32, block tile 128x64, K in 64-wide slabs, TDM
// double-buffered LDS (wave0 issues DMA, s_wait_tensorcnt retires it).
// ---------------------------------------------------------------------------
__global__ __launch_bounds__(256) void gemm_tdm(const __bf16* __restrict__ A,
                                                const __bf16* __restrict__ W,
                                                const float* __restrict__ bias,
                                                float* __restrict__ C,
                                                int M, int N, int K) {
  __shared__ __bf16 sA[2][128 * 72];
  __shared__ __bf16 sB[2][64 * 72];
  const int tid  = threadIdx.x;
  const int lane = tid & 31;
  const int wave = tid >> 5;
  const int wm = wave >> 1, wn = wave & 1;
  const int m0 = blockIdx.y * 128;
  const int n0 = blockIdx.x * 64;

  v8f acc[2][2];
#pragma unroll
  for (int i = 0; i < 2; ++i)
#pragma unroll
    for (int j = 0; j < 2; ++j) acc[i][j] = (v8f){};

  const int nSlab = K >> 6;
  if (wave == 0) {
    tdm_load_tile64(lds_off(&sA[0][0]), A + (size_t)m0 * K, 128, (unsigned)K);
    tdm_load_tile64(lds_off(&sB[0][0]), W + (size_t)n0 * K, 64,  (unsigned)K);
  }

  for (int i = 0; i < nSlab; ++i) {
    const int cur = i & 1;
    if (wave == 0) {
      if (i + 1 < nSlab) {
        const int k0 = (i + 1) << 6;
        const int nxt = cur ^ 1;
        tdm_load_tile64(lds_off(&sA[nxt][0]), A + (size_t)m0 * K + k0, 128, (unsigned)K);
        tdm_load_tile64(lds_off(&sB[nxt][0]), W + (size_t)n0 * K + k0, 64,  (unsigned)K);
        __builtin_amdgcn_s_wait_tensorcnt(2);   // slab i complete, slab i+1 in flight
      } else {
        __builtin_amdgcn_s_wait_tensorcnt(0);
      }
    }
    __syncthreads();

    const __bf16* pA = sA[cur];
    const __bf16* pB = sB[cur];
#pragma unroll
    for (int kk = 0; kk < 64; kk += 32) {
      v16bf a0 = load_a_frag(pA, 72, wm * 32,      kk, lane);
      v16bf a1 = load_a_frag(pA, 72, wm * 32 + 16, kk, lane);
      v16bf b0 = load_b_frag(pB, 72, wn * 32,      kk, lane);
      v16bf b1 = load_b_frag(pB, 72, wn * 32 + 16, kk, lane);
      acc[0][0] = __builtin_amdgcn_wmma_f32_16x16x32_bf16(false, a0, false, b0, (short)0, acc[0][0], false, false);
      acc[0][1] = __builtin_amdgcn_wmma_f32_16x16x32_bf16(false, a0, false, b1, (short)0, acc[0][1], false, false);
      acc[1][0] = __builtin_amdgcn_wmma_f32_16x16x32_bf16(false, a1, false, b0, (short)0, acc[1][0], false, false);
      acc[1][1] = __builtin_amdgcn_wmma_f32_16x16x32_bf16(false, a1, false, b1, (short)0, acc[1][1], false, false);
    }
    __syncthreads();
  }

#pragma unroll
  for (int i = 0; i < 2; ++i)
#pragma unroll
    for (int j = 0; j < 2; ++j) {
      const int col = n0 + wn * 32 + j * 16 + (lane & 15);
      const int rb  = m0 + wm * 32 + i * 16 + ((lane >> 4) << 3);
      const float bv = bias ? bias[col] : 0.0f;
#pragma unroll
      for (int r = 0; r < 8; ++r) C[(size_t)(rb + r) * N + col] = acc[i][j][r] + bv;
    }
}

// ---------------------------------------------------------------------------
// Elementwise / small kernels
// ---------------------------------------------------------------------------
__global__ void cvt_bf16_kernel(const float* __restrict__ s, __bf16* __restrict__ d, int n) {
  int i = blockIdx.x * 256 + threadIdx.x;
  if (i < n) d[i] = (__bf16)s[i];
}

__global__ void glu_kernel(const float* __restrict__ gp, float* __restrict__ xg) {
  int i = blockIdx.x * 256 + threadIdx.x;
  if (i >= MROWS * HID) return;
  int c = i % HID, bs = i / HID;
  float a = gp[(size_t)bs * 2 * HID + c];
  float g = gp[(size_t)bs * 2 * HID + HID + c];
  xg[i] = a * sigmoidf(g);
}

__global__ void router_kernel(const float* __restrict__ x, const float* __restrict__ rw,
                              const float* __restrict__ rb, float* __restrict__ hw) {
  int i = blockIdx.x * 256 + threadIdx.x;
  if (i >= MROWS * NHEADS) return;
  int h = i % NHEADS, bs = i / NHEADS;
  const float* xr = x + (size_t)bs * HID;
  const float* wr = rw + (size_t)h * HID;
  float a = rb[h];
  for (int k = 0; k < HID; ++k) a += xr[k] * wr[k];
  hw[i] = sigmoidf(a);
}

__global__ void conv_kernel(const float* __restrict__ in, float* __restrict__ out,
                            const float* __restrict__ cw, const float* __restrict__ cb,
                            int stage) {
  int i = blockIdx.x * 256 + threadIdx.x;
  if (i >= MROWS * HID) return;
  int cg = i % HID, bs = i / HID;
  int b = bs / SEQ, s = bs % SEQ;
  int head = cg >> 6, c = cg & 63;
  int d = c_dil[head][stage];
  const float* wr = cw + (size_t)(((head * 3 + stage) * HDIM + c)) * 4;
  float acc = cb[(head * 3 + stage) * HDIM + c];
  const float* col = in + (size_t)b * SEQ * HID + cg;
#pragma unroll
  for (int t = 0; t < 4; ++t) {
    int sp = s - (3 - t) * d;
    if (sp >= 0) acc += wr[t] * col[(size_t)sp * HID];
  }
  out[i] = in[i] + acc;
}

__global__ void headmul_kernel(const float* __restrict__ h, const float* __restrict__ hw,
                               __bf16* __restrict__ u) {
  int i = blockIdx.x * 256 + threadIdx.x;
  if (i >= MROWS * HID) return;
  int cg = i % HID, bs = i / HID;
  u[i] = (__bf16)(h[i] * hw[(size_t)bs * NHEADS + (cg >> 6)]);
}

__global__ void swishmul_kernel(const __bf16* __restrict__ u, const float* __restrict__ g1,
                                __bf16* __restrict__ v) {
  int i = blockIdx.x * 256 + threadIdx.x;
  if (i >= MROWS * HID) return;
  v[i] = (__bf16)((float)u[i] * sigmoidf(g1[i]));
}

// ---------------------------------------------------------------------------
// Neural memory: one workgroup per (batch, mem-head); sequential over 64 chunks
// of STRIDE=64; all chunk GEMMs via WMMA from dynamic LDS; fast-weight state
// kept as bf16 M^T[mem=128][d=64] in LDS.
// ---------------------------------------------------------------------------
#define OFF_WQ   0u
#define OFF_WK   9216u
#define OFF_WV   18432u
#define OFF_WO   36864u
#define OFF_MT   54272u
#define OFF_X    72704u
#define OFF_Q    81920u
#define OFF_KT   91136u
#define OFF_VT   100352u
#define OFF_R    118784u
#define OFF_G    136192u
#define OFF_GW   136448u
#define OFF_STAT 136704u
#define MEM_SMEM 136720u

__global__ __launch_bounds__(256) void mem_kernel(const float* __restrict__ xg,
                                                  const float* __restrict__ Wq,
                                                  const float* __restrict__ Wk,
                                                  const float* __restrict__ Wv,
                                                  const float* __restrict__ Wgw,
                                                  const float* __restrict__ Wgb,
                                                  const float* __restrict__ Wout,
                                                  float* __restrict__ h) {
  extern __shared__ char smem[];
  __bf16* sWq = (__bf16*)(smem + OFF_WQ);
  __bf16* sWk = (__bf16*)(smem + OFF_WK);
  __bf16* sWv = (__bf16*)(smem + OFF_WV);
  __bf16* sWo = (__bf16*)(smem + OFF_WO);
  __bf16* sMt = (__bf16*)(smem + OFF_MT);
  __bf16* sX  = (__bf16*)(smem + OFF_X);
  __bf16* sQ  = (__bf16*)(smem + OFF_Q);
  __bf16* sKT = (__bf16*)(smem + OFF_KT);
  __bf16* sVT = (__bf16*)(smem + OFF_VT);
  __bf16* sR  = (__bf16*)(smem + OFF_R);
  float*  sG  = (float*)(smem + OFF_G);
  float*  sGw = (float*)(smem + OFF_GW);
  float*  sSt = (float*)(smem + OFF_STAT);

  const int tid = threadIdx.x, lane = tid & 31, wave = tid >> 5;
  const int b = blockIdx.x >> 2, m = blockIdx.x & 3, head = 6 + m;

  for (int i = tid; i < 64 * 64; i += 256) {
    int n = i >> 6, k = i & 63;
    sWq[n * 72 + k] = (__bf16)Wq[(size_t)m * 4096 + i];
    sWk[n * 72 + k] = (__bf16)Wk[(size_t)m * 4096 + i];
  }
  for (int i = tid; i < 128 * 64; i += 256) {
    int n = i >> 6, k = i & 63;
    sWv[n * 72 + k] = (__bf16)Wv[(size_t)m * 8192 + i];
    sMt[n * 72 + k] = (__bf16)0.0f;
  }
  for (int i = tid; i < 64 * 128; i += 256) {
    int n = i >> 7, k = i & 127;
    sWo[n * 136 + k] = (__bf16)Wout[(size_t)m * 8192 + i];
  }
  if (tid < 64) sGw[tid] = Wgw[m * 64 + tid];
  __syncthreads();
  const float gb = Wgb[m];

  for (int c = 0; c < SEQ / 64; ++c) {
    const int s0 = c * 64;
    for (int i = tid; i < 64 * 64; i += 256) {
      int s = i >> 6, d = i & 63;
      sX[s * 72 + d] = (__bf16)xg[((size_t)b * SEQ + s0 + s) * HID + head * 64 + d];
    }
    if (tid < 64) {
      const float* row = xg + ((size_t)b * SEQ + s0 + tid) * HID + head * 64;
      float a = gb;
      for (int d = 0; d < 64; ++d) a += row[d] * sGw[d];
      sG[tid] = sigmoidf(a);
    }
    __syncthreads();
    if (tid == 0) {
      float a = 0.0f;
      for (int s = 0; s < 64; ++s) a += sG[s];
      sSt[0] = a * (1.0f / 64.0f);
    }
    __syncthreads();

    wave_gemm(sX, 72, sWq, 72, 64, 64, 64, wave, lane,
              [&](int r, int cc, float v) { sQ[r * 72 + cc] = (__bf16)v; });
    wave_gemm(sX, 72, sWk, 72, 64, 64, 64, wave, lane,
              [&](int r, int cc, float v) { sKT[cc * 72 + r] = (__bf16)(v * sG[r]); });
    wave_gemm(sX, 72, sWv, 72, 64, 128, 64, wave, lane,
              [&](int r, int cc, float v) { sVT[cc * 72 + r] = (__bf16)v; });
    __syncthreads();

    wave_gemm(sQ, 72, sMt, 72, 64, 128, 64, wave, lane,
              [&](int r, int cc, float v) { sR[r * 136 + cc] = (__bf16)v; });
    __syncthreads();

    const float decay = 1.0f - sSt[0];
    wave_gemm(sKT, 72, sVT, 72, 64, 128, 64, wave, lane,
              [&](int r, int cc, float v) {
                float old = (float)sMt[cc * 72 + r];
                sMt[cc * 72 + r] = (__bf16)(decay * old + v);
              });

    wave_gemm(sR, 136, sWo, 136, 64, 64, 128, wave, lane,
              [&](int r, int cc, float v) {
                h[((size_t)b * SEQ + s0 + r) * HID + head * 64 + cc] += v;
              });
    __syncthreads();
  }
}

// ---------------------------------------------------------------------------
// Launch
// ---------------------------------------------------------------------------
extern "C" void kernel_launch(void* const* d_in, const int* in_sizes, int n_in,
                              void* d_out, int out_size, void* d_ws, size_t ws_size,
                              hipStream_t stream) {
  (void)in_sizes; (void)n_in; (void)out_size; (void)ws_size;
  const float* x       = (const float*)d_in[0];
  const float* conv_w  = (const float*)d_in[1];
  const float* conv_b  = (const float*)d_in[2];
  const float* mem_Wq  = (const float*)d_in[3];
  const float* mem_Wk  = (const float*)d_in[4];
  const float* mem_Wv  = (const float*)d_in[5];
  const float* mem_Wgw = (const float*)d_in[6];
  const float* mem_Wgb = (const float*)d_in[7];
  const float* mem_Wo  = (const float*)d_in[8];
  const float* gate_w  = (const float*)d_in[9];
  const float* rout_w  = (const float*)d_in[10];
  const float* rout_b  = (const float*)d_in[11];
  const float* mixg_w  = (const float*)d_in[12];
  const float* mixg_b  = (const float*)d_in[13];
  const float* mix_w   = (const float*)d_in[14];
  const float* mix_b   = (const float*)d_in[15];
  float* out = (float*)d_out;

  char* wsb = (char*)d_ws;
  float*  buf0 = (float*)(wsb);                    // [16384,768]  x_gated
  float*  buf1 = (float*)(wsb + 50331648u);        // [16384,768]  conv/h
  float*  buf2 = (float*)(wsb + 100663296u);       // [16384,1536] gp / conv ping / g1
  float*  hw   = (float*)(wsb + 201326592u);       // [16384,12]
  __bf16* xubf = (__bf16*)(wsb + 202113024u);      // [16384,768] bf16: x, later u
  __bf16* vbf  = (__bf16*)(wsb + 227278848u);      // [16384,768] bf16: v
  __bf16* gwbf = (__bf16*)(wsb + 252444672u);      // gate_w bf16
  __bf16* mgbf = (__bf16*)(wsb + 254803968u);      // mixg_w bf16
  __bf16* mxbf = (__bf16*)(wsb + 255983616u);      // mix_w bf16

  const dim3 blk(256);
  const int nEl = MROWS * HID;            // 12582912
  const int gEl = (nEl + 255) / 256;

  // 0. one-time bf16 conversions (activations + big weights)
  cvt_bf16_kernel<<<gEl, blk, 0, stream>>>(x, xubf, nEl);
  cvt_bf16_kernel<<<(2 * HID * HID / 256) * 1 + 1, blk, 0, stream>>>(gate_w, gwbf, 2 * HID * HID);
  cvt_bf16_kernel<<<(HID * HID + 255) / 256, blk, 0, stream>>>(mixg_w, mgbf, HID * HID);
  cvt_bf16_kernel<<<(HID * HID + 255) / 256, blk, 0, stream>>>(mix_w, mxbf, HID * HID);

  // 1. gp = x @ gate_w^T   [16384,1536]
  gemm_tdm<<<dim3(2 * HID / 64, MROWS / 128), blk, 0, stream>>>(xubf, gwbf, nullptr, buf2,
                                                                MROWS, 2 * HID, HID);
  // 2. x_gated = gp_lo * sigmoid(gp_hi)
  glu_kernel<<<gEl, blk, 0, stream>>>(buf2, buf0);
  // 3. head_weights
  router_kernel<<<(MROWS * NHEADS + 255) / 256, blk, 0, stream>>>(x, rout_w, rout_b, hw);
  // 4-6. three serial dilated causal depthwise convs (h = h + dconv(h))
  conv_kernel<<<gEl, blk, 0, stream>>>(buf0, buf1, conv_w, conv_b, 0);
  conv_kernel<<<gEl, blk, 0, stream>>>(buf1, buf2, conv_w, conv_b, 1);
  conv_kernel<<<gEl, blk, 0, stream>>>(buf2, buf1, conv_w, conv_b, 2);
  // 7. fast-weight memory for heads 6..9 (adds into buf1)
  mem_kernel<<<dim3(BATCH * 4), blk, MEM_SMEM, stream>>>(buf0, mem_Wq, mem_Wk, mem_Wv,
                                                         mem_Wgw, mem_Wgb, mem_Wo, buf1);
  // 8. u = h * head_weights (bf16, reuses x's bf16 buffer)
  headmul_kernel<<<gEl, blk, 0, stream>>>(buf1, hw, xubf);
  // 9. g1 = u @ mixg_w^T + mixg_b
  gemm_tdm<<<dim3(HID / 64, MROWS / 128), blk, 0, stream>>>(xubf, mgbf, mixg_b, buf2,
                                                            MROWS, HID, HID);
  // 10. v = u * sigmoid(g1)
  swishmul_kernel<<<gEl, blk, 0, stream>>>(xubf, buf2, vbf);
  // 11. out = v @ mix_w^T + mix_b
  gemm_tdm<<<dim3(HID / 64, MROWS / 128), blk, 0, stream>>>(vbf, mxbf, mix_b, out,
                                                            MROWS, HID, HID);
}
